// TransformerVAE_56470230007974
// MI455X (gfx1250) — matrix operand order repaired and checked by reading, not compiled, not run on hardware
//
#include <hip/hip_runtime.h>

// ---------------------------------------------------------------------------
// Model constants
// ---------------------------------------------------------------------------
#define DMODEL 512
#define DFFN   2048
#define NH     8
#define NLAYER 6
#define DLATN  128
#define NVOCAB 1024
#define SEQE   128
#define NBATCH 8
#define NBAR   16
#define SEQD   1280
#define HD     64

static const int MENC = SEQE * NBATCH * NBAR;   // 16384 encoder tokens
static const int MDEC = SEQD * NBATCH;          // 10240 decoder tokens

// ---------------------------------------------------------------------------
// WMMA types (gfx1250, wave32)
// ---------------------------------------------------------------------------
typedef __attribute__((ext_vector_type(16))) __bf16 v16bf;
typedef __attribute__((ext_vector_type(8)))  float  v8f;

__device__ inline unsigned short f2bf(float f) {
    unsigned int u = __builtin_bit_cast(unsigned int, f);
    u += 0x7FFFu + ((u >> 16) & 1u);          // round-to-nearest-even
    return (unsigned short)(u >> 16);
}
__device__ inline unsigned int pack2bf(float a, float b) {
    unsigned int ua = __builtin_bit_cast(unsigned int, a);
    unsigned int ub = __builtin_bit_cast(unsigned int, b);
    ua += 0x7FFFu + ((ua >> 16) & 1u);
    ub += 0x7FFFu + ((ub >> 16) & 1u);
    return (ua >> 16) | (ub & 0xFFFF0000u);
}

// ---------------------------------------------------------------------------
// Generic batched GEMM:  C = op(alpha * A*B + bias + resid)
//   A : [M,K], row stride ldA (K contiguous)
//   B : bT=1 -> [N,K] K-contiguous (weights W[n,k]); bT=0 -> [K,N] N-contiguous
//   Batch z -> z0=z/Hdiv, z1=z%Hdiv; per-operand base += z0*S0 + z1*S1.
//   Tile: 128x64 C per block (256 thr = 8 waves), K step 64.
//   LDS staged pre-swizzled into WMMA fragment layout (bf16), so each
//   fragment is one contiguous 32B per-lane load (2x ds_load_b128).
// ---------------------------------------------------------------------------
#define TM 128
#define TN 64
#define TK 64

// fragment-layout LDS index (in u32 units): [tile][ksub][lane][8 dwords]
#define AIDX(mt, ks, ln) ((((mt)*2 + (ks))*32 + (ln))*8)
#define BIDX(nt, ks, ln) ((((nt)*2 + (ks))*32 + (ln))*8)

__global__ void __launch_bounds__(256)
wmma_gemm(const float* __restrict__ A, const float* __restrict__ Bm,
          const float* __restrict__ bias, const float* __restrict__ resid,
          float* __restrict__ C,
          long ldA, long ldB, long ldC, long ldR,
          int M, int N, int K, int Hdiv,
          long aS0, long aS1, long bS0, long bS1,
          long cS0, long cS1, long rS0, long rS1,
          float alpha, int bT, int relu)
{
    __shared__ __align__(32) unsigned int AsU[(TM/16) * 2 * 32 * 8];  // 16 KB
    __shared__ __align__(32) unsigned int BsU[(TN/16) * 2 * 32 * 8];  //  8 KB

    const int z  = blockIdx.z;
    const int z0 = z / Hdiv;
    const int z1 = z - z0 * Hdiv;
    A  += z0 * aS0 + z1 * aS1;
    Bm += z0 * bS0 + z1 * bS1;
    C  += z0 * cS0 + z1 * cS1;
    if (resid) resid += z0 * rS0 + z1 * rS1;

    const int m0 = blockIdx.y * TM;
    const int n0 = blockIdx.x * TN;
    const int t    = threadIdx.x;
    const int wave = t >> 5;
    const int lane = t & 31;
    const int lm   = lane & 15;
    const int lhi  = lane >> 4;

    v8f acc[4];
    #pragma unroll
    for (int i = 0; i < 4; ++i)
        #pragma unroll
        for (int r = 0; r < 8; ++r) acc[i][r] = 0.0f;

    for (int k0 = 0; k0 < K; k0 += TK) {
        // ---- stage A: 128 rows x 64 k, float4 -> packed bf16, swizzled ----
        #pragma unroll
        for (int i = 0; i < 8; ++i) {
            const int chunk = i * 256 + t;                 // 2048 quads
            const int r  = chunk >> 4;                     // 16 quads per row
            const int kc = (chunk & 15) * 4;
            const float4 f = *(const float4*)&A[(long)(m0 + r) * ldA + (k0 + kc)];
            const unsigned int p0 = pack2bf(f.x, f.y);
            const unsigned int p1 = pack2bf(f.z, f.w);
            const int ksub = kc >> 5;
            const int kk   = kc & 31;
            const int la   = (r & 15) + 16 * (kk >> 4);         // A fragment lane
            const int j0   = kk & 15;                           // fragment elem
            unsigned long long p = (unsigned long long)p0 |
                                   ((unsigned long long)p1 << 32);
            *(unsigned long long*)&AsU[AIDX(r >> 4, ksub, la) + (j0 >> 1)] = p;
        }
        // ---- stage B: 64 x 64 ----
        if (bT) {
            #pragma unroll
            for (int i = 0; i < 4; ++i) {
                const int chunk = i * 256 + t;             // 1024 quads
                const int n  = chunk >> 4;
                const int kc = (chunk & 15) * 4;
                const float4 f = *(const float4*)&Bm[(long)(n0 + n) * ldB + (k0 + kc)];
                const unsigned int p0 = pack2bf(f.x, f.y);
                const unsigned int p1 = pack2bf(f.z, f.w);
                const int ksub = kc >> 5;
                const int kk   = kc & 31;
                const int lb   = (n & 15) + 16 * (kk >> 4);     // B fragment lane
                const int j0   = kk & 15;
                unsigned long long p = (unsigned long long)p0 |
                                       ((unsigned long long)p1 << 32);
                *(unsigned long long*)&BsU[BIDX(n >> 4, ksub, lb) + (j0 >> 1)] = p;
            }
        } else {
            unsigned short* BsH = (unsigned short*)BsU;
            #pragma unroll
            for (int i = 0; i < 4; ++i) {
                const int chunk = i * 256 + t;             // 64 k x 16 n-quads
                const int kk64 = chunk >> 4;
                const int nc   = (chunk & 15) * 4;
                const float4 f = *(const float4*)&Bm[(long)(k0 + kk64) * ldB + (n0 + nc)];
                const int ksub = kk64 >> 5;
                const int kk   = kk64 & 31;
                const int j    = kk & 15;
                const int nt   = nc >> 4;
                const int base = BIDX(nt, ksub, 0) * 2 + j;     // ushort units
                const int l0   = (nc & 15) + 16 * (kk >> 4);
                BsH[base + (l0 + 0) * 16] = f2bf(f.x);
                BsH[base + (l0 + 1) * 16] = f2bf(f.y);
                BsH[base + (l0 + 2) * 16] = f2bf(f.z);
                BsH[base + (l0 + 3) * 16] = f2bf(f.w);
            }
        }
        // ---- prefetch next K-stage tiles (global_prefetch_b8) ----
        if (k0 + TK < K) {
            __builtin_prefetch(&A[(long)(m0 + (t >> 1)) * ldA + (k0 + TK) + (t & 1) * 32], 0, 0);
            if (t < 128) {
                if (bT)
                    __builtin_prefetch(&Bm[(long)(n0 + (t >> 1)) * ldB + (k0 + TK) + (t & 1) * 32], 0, 0);
                else
                    __builtin_prefetch(&Bm[(long)((k0 + TK) + (t >> 1)) * ldB + n0 + (t & 1) * 32], 0, 0);
            }
        }
        __syncthreads();

        // ---- compute: wave w owns rows w*16..+15; 2 ksubs x 4 ntiles ----
        #pragma unroll
        for (int ks = 0; ks < 2; ++ks) {
            const v16bf fa = *(const v16bf*)&AsU[AIDX(wave, ks, lane)];
            #pragma unroll
            for (int nt = 0; nt < 4; ++nt) {
                const v16bf fb = *(const v16bf*)&BsU[BIDX(nt, ks, lane)];
                acc[nt] = __builtin_amdgcn_wmma_f32_16x16x32_bf16(
                    false, fa, false, fb, (short)0, acc[nt], false, false);
            }
        }
        __syncthreads();
    }

    // ---- epilogue: alpha, bias, residual, relu ----
    const int rowb = m0 + wave * 16 + lhi * 8;   // C VGPR r -> row r (+8 hi lanes)
    #pragma unroll
    for (int nt = 0; nt < 4; ++nt) {
        const int col = n0 + nt * 16 + lm;
        const float bv = bias ? bias[col] : 0.0f;
        #pragma unroll
        for (int r = 0; r < 8; ++r) {
            long row = rowb + r;
            float v = acc[nt][r] * alpha + bv;
            if (resid) v += resid[row * ldR + col];
            if (relu)  v = v > 0.0f ? v : 0.0f;
            C[row * ldC + col] = v;
        }
    }
}

// ---------------------------------------------------------------------------
// Softmax over rows of [batch, S, S]; optional causal mask (add -1e9, t > s)
// ---------------------------------------------------------------------------
__global__ void __launch_bounds__(256)
softmax_rows(float* __restrict__ SC, int S, int causal)
{
    const long row = blockIdx.x;
    const int  s   = (int)(row % S);
    float* p = SC + row * (long)S;
    const int t = threadIdx.x;
    __shared__ float red[256];

    float mx = -3.0e38f;
    for (int j = t; j < S; j += 256) {
        float v = p[j];
        if (causal && j > s) v += -1.0e9f;
        mx = fmaxf(mx, v);
    }
    red[t] = mx; __syncthreads();
    for (int o = 128; o > 0; o >>= 1) { if (t < o) red[t] = fmaxf(red[t], red[t + o]); __syncthreads(); }
    mx = red[0]; __syncthreads();

    float sum = 0.0f;
    for (int j = t; j < S; j += 256) {
        float v = p[j];
        if (causal && j > s) v += -1.0e9f;
        v = __expf(v - mx);
        p[j] = v;
        sum += v;
    }
    red[t] = sum; __syncthreads();
    for (int o = 128; o > 0; o >>= 1) { if (t < o) red[t] += red[t + o]; __syncthreads(); }
    const float inv = 1.0f / red[0];
    for (int j = t; j < S; j += 256) p[j] *= inv;
}

// ---------------------------------------------------------------------------
// LayerNorm per row (D=512), eps=1e-5
// ---------------------------------------------------------------------------
__global__ void __launch_bounds__(256)
layernorm_rows(const float* __restrict__ X, const float* __restrict__ w,
               const float* __restrict__ b, float* __restrict__ Y, int Dm)
{
    const long row = blockIdx.x;
    const float* x = X + row * (long)Dm;
    float* y = Y + row * (long)Dm;
    const int t = threadIdx.x;
    __shared__ float red[256];

    float s = 0.0f;
    for (int j = t; j < Dm; j += 256) s += x[j];
    red[t] = s; __syncthreads();
    for (int o = 128; o > 0; o >>= 1) { if (t < o) red[t] += red[t + o]; __syncthreads(); }
    const float mean = red[0] / (float)Dm; __syncthreads();

    float v = 0.0f;
    for (int j = t; j < Dm; j += 256) { float d = x[j] - mean; v += d * d; }
    red[t] = v; __syncthreads();
    for (int o = 128; o > 0; o >>= 1) { if (t < o) red[t] += red[t + o]; __syncthreads(); }
    const float rstd = rsqrtf(red[0] / (float)Dm + 1e-5f);

    for (int j = t; j < Dm; j += 256) y[j] = (x[j] - mean) * rstd * w[j] + b[j];
}

// ---------------------------------------------------------------------------
// Token embedding * sqrt(D) + sinusoidal PE.  row = s*rowsPerS + n
// ---------------------------------------------------------------------------
__global__ void __launch_bounds__(128)
embed_kernel(const int* __restrict__ toks, const float* __restrict__ emb,
             float* __restrict__ X, int rowsPerS, float scale)
{
    const long row = blockIdx.x;
    const int  s   = (int)(row / rowsPerS);
    const int  tok = toks[row];
    const float* e = emb + (long)tok * DMODEL;
    float* x = X + row * (long)DMODEL;
    for (int j = threadIdx.x; j < DMODEL; j += blockDim.x) {
        int i2 = j & ~1;
        float div = __expf((float)i2 * (-9.2103403719761836f / (float)DMODEL));
        float a = (float)s * div;
        float pe = (j & 1) ? __cosf(a) : __sinf(a);
        x[j] = e[j] * scale + pe;
    }
}

// ---------------------------------------------------------------------------
// lat = eps * exp(0.5*logvar) + mu
// ---------------------------------------------------------------------------
__global__ void latent_kernel(const float* __restrict__ mu, const float* __restrict__ lv,
                              const float* __restrict__ eps, float* __restrict__ lat, int n)
{
    int i = blockIdx.x * 256 + threadIdx.x;
    if (i < n) lat[i] = eps[i] * __expf(0.5f * lv[i]) + mu[i];
}

// ---------------------------------------------------------------------------
// Bar-position scatter: seg[s,b,:] = valid ? lat[b, clip(bar_id)] : 0
// ---------------------------------------------------------------------------
__global__ void __launch_bounds__(128)
seg_kernel(const float* __restrict__ lat, const int* __restrict__ barpos,
           float* __restrict__ seg)
{
    const long row = blockIdx.x;           // s*NBATCH + b
    const int  s   = (int)(row / NBATCH);
    const int  b   = (int)(row % NBATCH);
    const int* bp = barpos + b * (NBAR + 1);
    int bid = 0;
    #pragma unroll
    for (int j = 1; j <= NBAR; ++j) bid += (s >= bp[j]) ? 1 : 0;
    const bool valid = (s >= bp[0]) && (bid < NBAR);
    int bc = bid; if (bc > NBAR - 1) bc = NBAR - 1; if (bc < 0) bc = 0;
    const float* src = lat + ((long)b * NBAR + bc) * DLATN;
    float* dst = seg + row * (long)DLATN;
    dst[threadIdx.x] = valid ? src[threadIdx.x] : 0.0f;
}

__global__ void add_kernel(float* __restrict__ y, const float* __restrict__ a, long n)
{
    long i = (long)blockIdx.x * 256 + threadIdx.x;
    if (i < n) y[i] += a[i];
}

// ---------------------------------------------------------------------------
// Host-side helpers
// ---------------------------------------------------------------------------
static inline void launch_gemm(hipStream_t st,
    const float* A, long ldA, const float* Bm, long ldB, int bT,
    const float* bias, const float* resid, long ldR,
    float* C, long ldC, int M, int N, int K,
    int batch, int Hdiv,
    long aS0, long aS1, long bS0, long bS1,
    long cS0, long cS1, long rS0, long rS1,
    float alpha, int relu)
{
    dim3 g(N / TN, M / TM, batch);
    wmma_gemm<<<g, dim3(256), 0, st>>>(A, Bm, bias, resid, C,
        ldA, ldB, ldC, ldR, M, N, K, Hdiv,
        aS0, aS1, bS0, bS1, cS0, cS1, rS0, rS1, alpha, bT, relu);
}

struct LayerP {
    const float *Wqkv, *bqkv, *Wo, *bo, *ln1w, *ln1b, *ln2w, *ln2b, *W1, *b1, *W2, *b2;
};

static void run_layer(hipStream_t st, const LayerP& p,
                      float* X, float* H1, float* T, float* O,
                      float* QKV, float* F1, float* SC,
                      int S, int Bn, int causal)
{
    const int M = S * Bn;
    // QKV projection: [M,3D] = X * Wqkv^T + bqkv
    launch_gemm(st, X, DMODEL, p.Wqkv, DMODEL, 1, p.bqkv, nullptr, 0,
                QKV, 3 * DMODEL, M, 3 * DMODEL, DMODEL,
                1, 1, 0, 0, 0, 0, 0, 0, 0, 0, 1.0f, 0);

    const long ldq  = (long)Bn * 3 * DMODEL;
    const int  bper = Bn / 8;                 // chunk b-dim 8 ways (SC scratch cap)
    for (int c = 0; c < 8; ++c) {
        const long boff  = (long)c * bper * 3 * DMODEL;
        const int  nb    = bper * NH;
        // scores[b,h,s,t] = q.k / sqrt(hd)
        launch_gemm(st, QKV + boff, ldq, QKV + boff + DMODEL, ldq, 1,
                    nullptr, nullptr, 0,
                    SC, S, S, S, HD,
                    nb, NH,
                    3 * DMODEL, HD, 3 * DMODEL, HD,
                    (long)NH * S * S, (long)S * S, 0, 0, 0.125f, 0);
        softmax_rows<<<dim3(nb * S), dim3(256), 0, st>>>(SC, S, causal);
        // O[s,b,h,:] = A.V
        launch_gemm(st, SC, S, QKV + boff + 2 * DMODEL, ldq, 0,
                    nullptr, nullptr, 0,
                    O + (long)c * bper * DMODEL, (long)Bn * DMODEL, S, HD, S,
                    nb, NH,
                    (long)NH * S * S, (long)S * S, 3 * DMODEL, HD,
                    DMODEL, HD, 0, 0, 1.0f, 0);
    }
    // T = O*Wo^T + bo + X  (residual fused)
    launch_gemm(st, O, DMODEL, p.Wo, DMODEL, 1, p.bo, X, DMODEL,
                T, DMODEL, M, DMODEL, DMODEL,
                1, 1, 0, 0, 0, 0, 0, 0, 0, 0, 1.0f, 0);
    layernorm_rows<<<dim3(M), dim3(256), 0, st>>>(T, p.ln1w, p.ln1b, H1, DMODEL);
    // FFN
    launch_gemm(st, H1, DMODEL, p.W1, DMODEL, 1, p.b1, nullptr, 0,
                F1, DFFN, M, DFFN, DMODEL,
                1, 1, 0, 0, 0, 0, 0, 0, 0, 0, 1.0f, 1);
    launch_gemm(st, F1, DFFN, p.W2, DFFN, 1, p.b2, H1, DMODEL,
                T, DMODEL, M, DMODEL, DFFN,
                1, 1, 0, 0, 0, 0, 0, 0, 0, 0, 1.0f, 0);
    layernorm_rows<<<dim3(M), dim3(256), 0, st>>>(T, p.ln2w, p.ln2b, X, DMODEL);
}

// ---------------------------------------------------------------------------
// Workspace arena (floats)
// ---------------------------------------------------------------------------
static const size_t MAXM     = 16384;
static const size_t OFF_X    = 0;
static const size_t OFF_H1   = OFF_X    + MAXM * DMODEL;
static const size_t OFF_T    = OFF_H1   + MAXM * DMODEL;
static const size_t OFF_O    = OFF_T    + MAXM * DMODEL;
static const size_t OFF_QKV  = OFF_O    + MAXM * DMODEL;
static const size_t OFF_F1   = OFF_QKV  + MAXM * 3 * DMODEL;
static const size_t OFF_SC   = OFF_F1   + MAXM * DFFN;
static const size_t SC_ELTS  = (size_t)NH * SEQD * SEQD;     // 13.1M (dec chunk) > enc chunk
static const size_t OFF_SEG  = OFF_SC   + SC_ELTS;
static const size_t OFF_SEGP = OFF_SEG  + (size_t)SEQD * NBATCH * DLATN;
static const size_t OFF_LAT  = OFF_SEGP + (size_t)SEQD * NBATCH * DMODEL;

extern "C" void kernel_launch(void* const* d_in, const int* in_sizes, int n_in,
                              void* d_out, int out_size, void* d_ws, size_t ws_size,
                              hipStream_t stream)
{
    (void)in_sizes; (void)n_in; (void)out_size; (void)ws_size;

    const int*   enc_inp = (const int*)d_in[0];
    const int*   dec_inp = (const int*)d_in[1];
    const int*   barpos  = (const int*)d_in[2];
    const float* eps     = (const float*)d_in[3];
    const float* tok_emb = (const float*)d_in[4];
    const float* encB[12]; for (int i = 0; i < 12; ++i) encB[i] = (const float*)d_in[5 + i];
    const float* decB[12]; for (int i = 0; i < 12; ++i) decB[i] = (const float*)d_in[17 + i];
    const float* fc_mu_w = (const float*)d_in[29];
    const float* fc_mu_b = (const float*)d_in[30];
    const float* fc_lv_w = (const float*)d_in[31];
    const float* fc_lv_b = (const float*)d_in[32];
    const float* segpj_w = (const float*)d_in[33];
    const float* outp_w  = (const float*)d_in[34];
    const float* outp_b  = (const float*)d_in[35];

    float* out = (float*)d_out;
    float* mu_out = out;                       // [128,128]
    float* lv_out = out + 16384;               // [128,128]
    float* logits = out + 32768;               // [1280,8,1024]

    float* ws   = (float*)d_ws;
    float* X    = ws + OFF_X;
    float* H1   = ws + OFF_H1;
    float* T    = ws + OFF_T;
    float* O    = ws + OFF_O;
    float* QKV  = ws + OFF_QKV;
    float* F1   = ws + OFF_F1;
    float* SC   = ws + OFF_SC;
    float* SEG  = ws + OFF_SEG;
    float* SEGP = ws + OFF_SEGP;
    float* LAT  = ws + OFF_LAT;

    // per-layer parameter strides for the 12 arrays
    static const long PO[12] = {
        3L * DMODEL * DMODEL, 3L * DMODEL, (long)DMODEL * DMODEL, DMODEL,
        DMODEL, DMODEL, DMODEL, DMODEL,
        (long)DFFN * DMODEL, DFFN, (long)DMODEL * DFFN, DMODEL };

    const float SCALE = 22.627416997969522f;   // sqrt(512)

    // ===================== Encoder =====================
    embed_kernel<<<dim3(MENC), dim3(128), 0, stream>>>(enc_inp, tok_emb, X,
                                                       NBATCH * NBAR, SCALE);
    for (int i = 0; i < NLAYER; ++i) {
        LayerP p = { encB[0] + i * PO[0], encB[1] + i * PO[1], encB[2] + i * PO[2],
                     encB[3] + i * PO[3], encB[4] + i * PO[4], encB[5] + i * PO[5],
                     encB[6] + i * PO[6], encB[7] + i * PO[7], encB[8] + i * PO[8],
                     encB[9] + i * PO[9], encB[10] + i * PO[10], encB[11] + i * PO[11] };
        run_layer(stream, p, X, H1, T, O, QKV, F1, SC, SEQE, NBATCH * NBAR, 0);
    }
    // hidden = enc_x[0] -> first 128 rows of X.  mu / logvar straight into d_out.
    launch_gemm(stream, X, DMODEL, fc_mu_w, DMODEL, 1, fc_mu_b, nullptr, 0,
                mu_out, DLATN, 128, DLATN, DMODEL,
                1, 1, 0, 0, 0, 0, 0, 0, 0, 0, 1.0f, 0);
    launch_gemm(stream, X, DMODEL, fc_lv_w, DMODEL, 1, fc_lv_b, nullptr, 0,
                lv_out, DLATN, 128, DLATN, DMODEL,
                1, 1, 0, 0, 0, 0, 0, 0, 0, 0, 1.0f, 0);
    latent_kernel<<<dim3(64), dim3(256), 0, stream>>>(mu_out, lv_out, eps, LAT, 16384);

    // ===================== Decoder =====================
    embed_kernel<<<dim3(MDEC), dim3(128), 0, stream>>>(dec_inp, tok_emb, X, NBATCH, SCALE);
    seg_kernel<<<dim3(MDEC), dim3(128), 0, stream>>>(LAT, barpos, SEG);
    launch_gemm(stream, SEG, DLATN, segpj_w, DLATN, 1, nullptr, nullptr, 0,
                SEGP, DMODEL, MDEC, DMODEL, DLATN,
                1, 1, 0, 0, 0, 0, 0, 0, 0, 0, 1.0f, 0);
    for (int i = 0; i < NLAYER; ++i) {
        add_kernel<<<dim3((unsigned)((long)MDEC * DMODEL / 256)), dim3(256), 0, stream>>>(
            X, SEGP, (long)MDEC * DMODEL);
        LayerP p = { decB[0] + i * PO[0], decB[1] + i * PO[1], decB[2] + i * PO[2],
                     decB[3] + i * PO[3], decB[4] + i * PO[4], decB[5] + i * PO[5],
                     decB[6] + i * PO[6], decB[7] + i * PO[7], decB[8] + i * PO[8],
                     decB[9] + i * PO[9], decB[10] + i * PO[10], decB[11] + i * PO[11] };
        run_layer(stream, p, X, H1, T, O, QKV, F1, SC, SEQD, NBATCH, 1);
    }
    // logits = X * out_proj_w^T + b  -> d_out[32768:]
    launch_gemm(stream, X, DMODEL, outp_w, DMODEL, 1, outp_b, nullptr, 0,
                logits, NVOCAB, MDEC, NVOCAB, DMODEL,
                1, 1, 0, 0, 0, 0, 0, 0, 0, 0, 1.0f, 0);
}
